// Attn_65335042507385
// MI455X (gfx1250) — matrix operand order repaired and checked by reading, not compiled, not run on hardware
//
#include <hip/hip_runtime.h>
#include <hip/hip_bf16.h>

// ---------------------------------------------------------------------------
// Problem constants (match reference)
// ---------------------------------------------------------------------------
static constexpr int Bc  = 2;
static constexpr int Tc  = 2048;
static constexpr int DMc = 1024;
static constexpr int Hc  = 16;
static constexpr int Dc  = 64;      // head dim
static constexpr float ROPE_BASE = 10000.0f;

// ---------------------------------------------------------------------------
// WMMA fragment types (CDNA5 / gfx1250, wave32)
// ---------------------------------------------------------------------------
typedef __bf16 v16bf __attribute__((ext_vector_type(16)));
typedef float  v8f   __attribute__((ext_vector_type(8)));
typedef unsigned int u32x4 __attribute__((ext_vector_type(4)));
typedef int          i32x4 __attribute__((ext_vector_type(4)));
typedef int          i32x8 __attribute__((ext_vector_type(8)));

union FragAB { v16bf v; unsigned short u[16]; float4 q2[2]; };
union FragC  { v8f   v; float f[8]; };

__device__ __forceinline__ unsigned short f2bf(float x) {
    unsigned int u = __float_as_uint(x);
    u += 0x7FFFu + ((u >> 16) & 1u);       // round-to-nearest-even
    return (unsigned short)(u >> 16);
}

// A-matrix 16x32 bf16 fragment (ISA 7.12.2): lanes 0-15 hold row M=lane,
// K in {0..7, 16..23}; lanes 16-31 hold row M=lane-16, K in {8..15, 24..31}.
__device__ __forceinline__ v16bf load_frag_a(const unsigned short* base, int ld) {
    const int lane = threadIdx.x & 31;
    const int row  = lane & 15;
    const int kb   = (lane >> 4) * 8;                 // 0 or 8
    const unsigned short* p = base + row * ld + kb;
    FragAB f;
    f.q2[0] = *(const float4*)(p);                    // K kb .. kb+7
    f.q2[1] = *(const float4*)(p + 16);               // K kb+16 .. kb+23
    return f.v;
}

// B-matrix 32x16 bf16 fragment: lane n<16 holds column N=n, K in {0..7, 16..23};
// lane n>=16 holds column N=n-16, K in {8..15, 24..31}.  `base` points at the
// tile such that column n is the ld-strided row n (i.e. K^T style storage).
__device__ __forceinline__ v16bf load_frag_b(const unsigned short* base, int ld) {
    const int lane = threadIdx.x & 31;
    const int col  = lane & 15;
    const int kb   = (lane >> 4) * 8;                 // 0 or 8
    const unsigned short* p = base + col * ld + kb;
    FragAB f;
    f.q2[0] = *(const float4*)(p);                    // K kb .. kb+7
    f.q2[1] = *(const float4*)(p + 16);               // K kb+16 .. kb+23
    return f.v;
}

__device__ __forceinline__ v8f wmma_bf16(v16bf a, v16bf b, v8f c) {
    // D(16x16,f32) = A(16x32,bf16) * B(32x16,bf16) + C
    return __builtin_amdgcn_wmma_f32_16x16x32_bf16(
        /*neg_a=*/false, a, /*neg_b=*/false, b,
        /*c_mod=*/(short)0, c, /*reuse_a=*/false, /*reuse_b=*/false);
}

// ---------------------------------------------------------------------------
// Tensor Data Mover: 2D tile load global -> LDS (D# per ISA cdna5 §8.3/8.4)
//   data_size = 2 bytes (bf16). tile_w elements contiguous per row,
//   tile_h rows, row stride = stride_elts.
//   Toolchain uses the 6-arg builtin: (g0, g1, g2, g3, g4, cpol).
// ---------------------------------------------------------------------------
__device__ __forceinline__ void tdm_load_2d(unsigned lds_off,
                                            const unsigned short* gptr,
                                            unsigned tile_w, unsigned tile_h,
                                            unsigned stride_elts,
                                            unsigned dim0, unsigned dim1) {
    const unsigned long long ga = (unsigned long long)(uintptr_t)gptr;
    u32x4 g0;
    g0.x = 1u;                                        // count=1, is_restore=0
    g0.y = lds_off;                                   // lds_addr (bytes)
    g0.z = (unsigned)(ga & 0xFFFFFFFFu);              // global_addr[31:0]
    g0.w = (unsigned)((ga >> 32) & 0x01FFFFFFu)       // global_addr[56:32]
           | (2u << 30);                              // type = 2 ("image")
    i32x8 g1;
    g1[0] = (int)(1u << 16);                          // wg_mask=0, data_size=1 (2B)
    g1[1] = (int)((dim0 & 0xFFFFu) << 16);            // tensor_dim0[15:0]
    g1[2] = (int)(((dim0 >> 16) & 0xFFFFu) | ((dim1 & 0xFFFFu) << 16));
    g1[3] = (int)(((dim1 >> 16) & 0xFFFFu) | ((tile_w & 0xFFFFu) << 16));
    g1[4] = (int)(tile_h & 0xFFFFu);                  // tile_dim1; tile_dim2=0
    g1[5] = (int)stride_elts;                         // tensor_dim0_stride[31:0]
    g1[6] = 0;                                        // stride hi / dim1_stride lo
    g1[7] = 0;
    const i32x4 z4 = {0, 0, 0, 0};
    const i32x8 z8 = {0, 0, 0, 0, 0, 0, 0, 0};
    __builtin_amdgcn_tensor_load_to_lds(g0, g1, z4, z4, z8, 0);
}

// ---------------------------------------------------------------------------
// Kernel 1: f32 -> bf16 conversion
// ---------------------------------------------------------------------------
__global__ void cvt_bf16_kernel(const float* __restrict__ in,
                                unsigned short* __restrict__ out, int n) {
    int i = blockIdx.x * blockDim.x + threadIdx.x;
    if (i < n) out[i] = f2bf(in[i]);
}

// ---------------------------------------------------------------------------
// Kernel 2: RoPE cos/sin cache [T][D/2]
// ---------------------------------------------------------------------------
__global__ void rope_cache_kernel(float* __restrict__ cosT,
                                  float* __restrict__ sinT) {
    int i = blockIdx.x * blockDim.x + threadIdx.x;
    if (i >= Tc * (Dc / 2)) return;
    int t = i >> 5;                 // D/2 == 32
    int j = i & 31;
    float inv = __powf(ROPE_BASE, -(2.0f * (float)j) / (float)Dc);
    float f   = (float)t * inv;
    cosT[i] = __cosf(f);
    sinT[i] = __sinf(f);
}

// ---------------------------------------------------------------------------
// Kernel 3: QKV GEMM (x @ Wqkv^T) + RoPE epilogue
//   one wave computes a 32x64 output tile (8 WMMAs per K-step of 32, B frags
//   shared between the two row halves). q,k -> [B,H,T,D] bf16 (RoPE applied);
//   v -> [B,H,D,T] bf16 (transposed for the PV WMMA B-operand).
// ---------------------------------------------------------------------------
__global__ __launch_bounds__(32)
void qkv_gemm_rope_kernel(const unsigned short* __restrict__ xb,
                          const unsigned short* __restrict__ wb,
                          const float* __restrict__ cosT,
                          const float* __restrict__ sinT,
                          unsigned short* __restrict__ qo,
                          unsigned short* __restrict__ ko,
                          unsigned short* __restrict__ vto) {
    const int row0 = blockIdx.x * 32;       // 0 .. B*T-1 rows
    const int n0   = blockIdx.y * 64;       // 0 .. 3*DM-1 cols
    const int lane = threadIdx.x & 31;
    const int half = lane >> 4;
    const int colL = lane & 15;

    FragC acc[2][4];
#pragma unroll
    for (int r = 0; r < 2; ++r)
#pragma unroll
        for (int j = 0; j < 4; ++j)
#pragma unroll
            for (int i = 0; i < 8; ++i) acc[r][j].f[i] = 0.0f;

    const unsigned short* aRow = xb + (size_t)row0 * DMc;
    for (int k = 0; k < DMc; k += 32) {
        v16bf a0 = load_frag_a(aRow + k, DMc);
        v16bf a1 = load_frag_a(aRow + (size_t)16 * DMc + k, DMc);
#pragma unroll
        for (int j = 0; j < 4; ++j) {
            v16bf b = load_frag_b(wb + (size_t)(n0 + j * 16) * DMc + k, DMc);
            acc[0][j].v = wmma_bf16(a0, b, acc[0][j].v);
            acc[1][j].v = wmma_bf16(a1, b, acc[1][j].v);
        }
        if (k + 32 < DMc) {
            __builtin_prefetch(aRow + k + 32, 0, 3);
            __builtin_prefetch(aRow + (size_t)16 * DMc + k + 32, 0, 3);
        }
    }

    __shared__ float sT[32][64 + 4];
#pragma unroll
    for (int r = 0; r < 2; ++r)
#pragma unroll
        for (int j = 0; j < 4; ++j)
#pragma unroll
            for (int i = 0; i < 8; ++i)
                sT[r * 16 + i + half * 8][j * 16 + colL] = acc[r][j].f[i];
    __syncthreads();

    const int region = n0 >> 10;            // 0=q 1=k 2=v
    const int cbase  = n0 & 1023;
    // 32 rows * 32 even/odd pairs
    for (int p = lane; p < 1024; p += 32) {
        const int row = p >> 5;
        const int pr  = p & 31;
        const int ld0 = pr * 2;
        const int gr  = row0 + row;
        const int b   = gr >> 11;           // / T
        const int t   = gr & 2047;          // % T
        const int c   = cbase + ld0;
        const int h   = c >> 6;
        const int d   = c & 63;
        const float x1 = sT[row][ld0];
        const float x2 = sT[row][ld0 + 1];
        if (region < 2) {
            const int j = d >> 1;
            const float cs = cosT[t * 32 + j];
            const float sn = sinT[t * 32 + j];
            unsigned short* dst = (region == 0) ? qo : ko;
            size_t base = (((size_t)(b * Hc + h) * Tc) + t) * Dc + d;
            dst[base]     = f2bf(x1 * cs - x2 * sn);
            dst[base + 1] = f2bf(x1 * sn + x2 * cs);
        } else {
            size_t base = ((size_t)(b * Hc + h) * Dc + d) * Tc + t;
            vto[base]      = f2bf(x1);
            vto[base + Tc] = f2bf(x2);
        }
    }
}

// ---------------------------------------------------------------------------
// Kernel 4: causal flash attention; one wave per (b, h, 16-row q tile).
//   K/V tiles staged global->LDS by the Tensor Data Mover, double-buffered so
//   the DMA for tile i+1 overlaps WMMA work on tile i (TENSORcnt pipeline).
//   scores = (Q Kt)/8, online softmax, y = P V.  y -> [B,T,DM] bf16.
// ---------------------------------------------------------------------------
__global__ __launch_bounds__(32)
void flash_attn_kernel(const unsigned short* __restrict__ qb,
                       const unsigned short* __restrict__ kbuf,
                       const unsigned short* __restrict__ vtb,
                       unsigned short* __restrict__ yb) {
    const int q0   = blockIdx.x * 16;
    const int h    = blockIdx.y;
    const int b    = blockIdx.z;
    const int lane = threadIdx.x & 31;
    const int half = lane >> 4;
    const int colL = lane & 15;

    const unsigned short* qB = qb   + (((size_t)(b * Hc + h) * Tc) + q0) * Dc;
    const unsigned short* kB = kbuf + ((size_t)(b * Hc + h) * Tc) * Dc;
    const unsigned short* vB = vtb  + ((size_t)(b * Hc + h) * Dc) * Tc;

    // Q tile 16x64 = two chained A fragments (resident whole kernel)
    const v16bf qf0 = load_frag_a(qB, Dc);
    const v16bf qf1 = load_frag_a(qB + 32, Dc);

    FragC yacc[4];
#pragma unroll
    for (int j = 0; j < 4; ++j)
#pragma unroll
        for (int i = 0; i < 8; ++i) yacc[j].f[i] = 0.0f;

    float m_r = -INFINITY, l_r = 0.0f;      // live in lanes 0..15 (row stats)

    __shared__ __align__(16) unsigned short sK[2][32 * 64]; // keys x D
    __shared__ __align__(16) unsigned short sV[2][64 * 32]; // dims x keys
    __shared__ float          sS[16][32];
    __shared__ unsigned short sP[16][32];
    __shared__ float          sScale[16];

    const unsigned ldsK0 = (unsigned)(uintptr_t)&sK[0][0];  // low 32 bits of a
    const unsigned ldsK1 = (unsigned)(uintptr_t)&sK[1][0];  // generic LDS ptr
    const unsigned ldsV0 = (unsigned)(uintptr_t)&sV[0][0];  // == LDS offset
    const unsigned ldsV1 = (unsigned)(uintptr_t)&sV[1][0];

    const float rs = 0.125f;                // 1/sqrt(64)
    const int   kend = q0 + 16;             // causal bound (exclusive)
    const int   nt   = (kend + 31) >> 5;    // number of 32-key tiles

    // prologue: TDM-load tile 0 into buffer 0 (K: 32x64 rows, V: 64x32 rows)
    tdm_load_2d(ldsK0, kB, /*w*/64, /*h*/32, /*stride*/Dc, 64u, (unsigned)Tc);
    tdm_load_2d(ldsV0, vB, /*w*/32, /*h*/64, /*stride*/Tc, (unsigned)Tc, 64u);

    int buf = 0;
    for (int it = 0; it < nt; ++it) {
        const int kt = it << 5;
        if (it + 1 < nt) {
            const int ktn = kt + 32;
            tdm_load_2d(buf ? ldsK0 : ldsK1, kB + (size_t)ktn * Dc,
                        64, 32, Dc, 64u, (unsigned)Tc);
            tdm_load_2d(buf ? ldsV0 : ldsV1, vB + ktn,
                        32, 64, Tc, (unsigned)Tc, 64u);
            __builtin_amdgcn_s_wait_tensorcnt(2);   // tile `it` landed in LDS
        } else {
            __builtin_amdgcn_s_wait_tensorcnt(0);
        }
        const unsigned short* kT = &sK[buf][0];     // ld = 64
        const unsigned short* vT = &sV[buf][0];     // ld = 32

        // ---- scores: two 16-key groups, K-dim 64 = 2 chained WMMAs each ----
#pragma unroll
        for (int n = 0; n < 2; ++n) {
            FragC sacc;
#pragma unroll
            for (int i = 0; i < 8; ++i) sacc.f[i] = 0.0f;
            const unsigned short* kTile = kT + n * 16 * 64;
            sacc.v = wmma_bf16(qf0, load_frag_b(kTile,      64), sacc.v);
            sacc.v = wmma_bf16(qf1, load_frag_b(kTile + 32, 64), sacc.v);
#pragma unroll
            for (int i = 0; i < 8; ++i) {
                const int row = i + half * 8;
                const int key = kt + n * 16 + colL;
                float sv = sacc.f[i] * rs;
                if (key > q0 + row) sv = -INFINITY;   // causal mask
                sS[row][n * 16 + colL] = sv;
            }
        }
        __syncthreads();

        // ---- online softmax: lane r owns query row r ----
        if (lane < 16) {
            const int r = lane;
            float mx = m_r;
#pragma unroll
            for (int c = 0; c < 32; ++c) mx = fmaxf(mx, sS[r][c]);
            const float corr = __expf(m_r - mx);
            float s = 0.0f;
#pragma unroll
            for (int c = 0; c < 32; ++c) {
                const float e = __expf(sS[r][c] - mx);
                s += e;
                sP[r][c] = f2bf(e);
            }
            l_r = l_r * corr + s;
            m_r = mx;
            sScale[r] = corr;
        }
        __syncthreads();

        // ---- rescale accumulators by per-row correction ----
        float sc[8];
#pragma unroll
        for (int i = 0; i < 8; ++i) sc[i] = sScale[i + half * 8];
#pragma unroll
        for (int j = 0; j < 4; ++j)
#pragma unroll
            for (int i = 0; i < 8; ++i) yacc[j].f[i] *= sc[i];

        // ---- y += P(16x32) @ V(32x64): 4 WMMAs, V fragments from LDS ----
        const v16bf pf = load_frag_a(&sP[0][0], 32);
#pragma unroll
        for (int j = 0; j < 4; ++j) {
            v16bf vf = load_frag_b(vT + j * 16 * 32, 32);
            yacc[j].v = wmma_bf16(pf, vf, yacc[j].v);
        }
        __syncthreads();                    // protect sS/sP for next tile
        buf ^= 1;
    }

    if (lane < 16) sScale[lane] = 1.0f / l_r;
    __syncthreads();
    float inv[8];
#pragma unroll
    for (int i = 0; i < 8; ++i) inv[i] = sScale[i + half * 8];

    // store y [B,T,DM] bf16
#pragma unroll
    for (int j = 0; j < 4; ++j)
#pragma unroll
        for (int i = 0; i < 8; ++i) {
            const int row = i + half * 8;
            const int t   = q0 + row;
            yb[((size_t)(b * Tc + t) * DMc) + h * Dc + j * 16 + colL] =
                f2bf(yacc[j].f[i] * inv[i]);
        }
}

// ---------------------------------------------------------------------------
// Kernel 5: output projection (y @ Wproj^T), 32x64 tile per wave,
//   f32 result written straight from C fragments.
// ---------------------------------------------------------------------------
__global__ __launch_bounds__(32)
void proj_gemm_kernel(const unsigned short* __restrict__ yb,
                      const unsigned short* __restrict__ wb,
                      float* __restrict__ out) {
    const int row0 = blockIdx.x * 32;
    const int n0   = blockIdx.y * 64;
    const int lane = threadIdx.x & 31;
    const int half = lane >> 4;
    const int colL = lane & 15;

    FragC acc[2][4];
#pragma unroll
    for (int r = 0; r < 2; ++r)
#pragma unroll
        for (int j = 0; j < 4; ++j)
#pragma unroll
            for (int i = 0; i < 8; ++i) acc[r][j].f[i] = 0.0f;

    const unsigned short* aRow = yb + (size_t)row0 * DMc;
    for (int k = 0; k < DMc; k += 32) {
        v16bf a0 = load_frag_a(aRow + k, DMc);
        v16bf a1 = load_frag_a(aRow + (size_t)16 * DMc + k, DMc);
#pragma unroll
        for (int j = 0; j < 4; ++j) {
            v16bf b = load_frag_b(wb + (size_t)(n0 + j * 16) * DMc + k, DMc);
            acc[0][j].v = wmma_bf16(a0, b, acc[0][j].v);
            acc[1][j].v = wmma_bf16(a1, b, acc[1][j].v);
        }
        if (k + 32 < DMc) {
            __builtin_prefetch(aRow + k + 32, 0, 3);
            __builtin_prefetch(aRow + (size_t)16 * DMc + k + 32, 0, 3);
        }
    }

#pragma unroll
    for (int r = 0; r < 2; ++r)
#pragma unroll
        for (int j = 0; j < 4; ++j)
#pragma unroll
            for (int i = 0; i < 8; ++i)
                out[(size_t)(row0 + r * 16 + i + half * 8) * DMc
                    + n0 + j * 16 + colL] = acc[r][j].f[i];
}

// ---------------------------------------------------------------------------
// Launch
// ---------------------------------------------------------------------------
extern "C" void kernel_launch(void* const* d_in, const int* in_sizes, int n_in,
                              void* d_out, int out_size, void* d_ws, size_t ws_size,
                              hipStream_t stream) {
    (void)in_sizes; (void)n_in; (void)out_size; (void)ws_size;

    const float* x     = (const float*)d_in[0];   // [B,T,DM]
    const float* Wqkv  = (const float*)d_in[1];   // [3*DM, DM]
    const float* Wproj = (const float*)d_in[2];   // [DM, DM]
    float*       out   = (float*)d_out;           // [B,T,DM]

    char* ws = (char*)d_ws;
    const size_t nX  = (size_t)Bc * Tc * DMc;       // 4,194,304
    const size_t nWq = (size_t)3 * DMc * DMc;       // 3,145,728
    const size_t nWp = (size_t)DMc * DMc;           // 1,048,576
    const size_t nQ  = (size_t)Bc * Hc * Tc * Dc;   // 4,194,304

    unsigned short* xb     = (unsigned short*)(ws);
    unsigned short* wqkvb  = (unsigned short*)(ws + 2 * nX);
    unsigned short* wprojb = (unsigned short*)(ws + 2 * (nX + nWq));
    unsigned short* qb     = (unsigned short*)(ws + 2 * (nX + nWq + nWp));
    unsigned short* kb     = (unsigned short*)(ws + 2 * (nX + nWq + nWp + nQ));
    unsigned short* vtb    = (unsigned short*)(ws + 2 * (nX + nWq + nWp + 2 * nQ));
    unsigned short* ybuf   = (unsigned short*)(ws + 2 * (nX + nWq + nWp + 3 * nQ));
    float*          cosT   = (float*)(ws + 2 * (nX + nWq + nWp + 4 * nQ));
    float*          sinT   = cosT + (size_t)Tc * (Dc / 2);

    // 1) convert inputs to bf16
    cvt_bf16_kernel<<<dim3((unsigned)((nX  + 255) / 256)), 256, 0, stream>>>(x,     xb,     (int)nX);
    cvt_bf16_kernel<<<dim3((unsigned)((nWq + 255) / 256)), 256, 0, stream>>>(Wqkv,  wqkvb,  (int)nWq);
    cvt_bf16_kernel<<<dim3((unsigned)((nWp + 255) / 256)), 256, 0, stream>>>(Wproj, wprojb, (int)nWp);

    // 2) RoPE tables
    rope_cache_kernel<<<dim3((Tc * (Dc / 2) + 255) / 256), 256, 0, stream>>>(cosT, sinT);

    // 3) QKV projection + RoPE  (grid: 128 row tiles x 48 col tiles)
    qkv_gemm_rope_kernel<<<dim3(Bc * Tc / 32, 3 * DMc / 64), 32, 0, stream>>>(
        xb, wqkvb, cosT, sinT, qb, kb, vtb);

    // 4) flash attention  (grid: 128 q-tiles x 16 heads x 2 batch)
    flash_attn_kernel<<<dim3(Tc / 16, Hc, Bc), 32, 0, stream>>>(qb, kb, vtb, ybuf);

    // 5) output projection
    proj_gemm_kernel<<<dim3(Bc * Tc / 32, DMc / 64), 32, 0, stream>>>(ybuf, wprojb, out);
}
